// ListMLELoss_84387517432679
// MI455X (gfx1250) — compile-verified, matrix operand order checked by reading, not run on hardware
//
#include <hip/hip_runtime.h>
#include <hip/hip_bf16.h>

typedef __attribute__((ext_vector_type(2)))  float    v2f;
typedef __attribute__((ext_vector_type(8)))  float    v8f;
typedef __attribute__((ext_vector_type(16))) _Float16 v16h;

static constexpr int   LROW  = 2048;        // row length
static constexpr int   TPB   = 256;         // 8 wave32s per workgroup
static constexpr int   EPT   = LROW / TPB;  // 8 elements per thread
static constexpr float kNEG  = -1.0e9f;     // reference's stand-in for -inf
static constexpr float kNINF = -3.0e38f;    // scan identity (exp() underflows to 0)

__device__ __forceinline__ float lse2(float a, float b) {
    float m = fmaxf(a, b);
    return m + __logf(__expf(a - m) + __expf(b - m));
}

// One workgroup per row: hybrid LDS/register bitonic sort, suffix
// logcumsumexp, WMMA row-sum.
__global__ __launch_bounds__(TPB) void listmle_row_kernel(
    const float* __restrict__ preds,
    const float* __restrict__ labels,
    const int*   __restrict__ mask,
    float*       __restrict__ rowOut,   // det: [B*2] (loss, valid); atomic: [2]
    int atomicMode)
{
    __shared__ float sKey[LROW];   // sort keys, later reused as `terms`
    __shared__ float sVal[LROW];   // pm = masked preds (valid pred or kNEG)
    __shared__ float sScan[TPB];
    __shared__ float sPart[2];
    __shared__ int   sK;

    const int    row  = blockIdx.x;
    const int    tid  = threadIdx.x;
    const size_t base = (size_t)row * LROW;

    // ---- vectorized load + mask (global_load_b128) -------------------------
    {
        const float4* p4 = (const float4*)(preds  + base);
        const float4* l4 = (const float4*)(labels + base);
        const int4*   m4 = (const int4*)  (mask   + base);
        for (int i = tid; i < LROW / 4; i += TPB) {   // 2 iterations
            float4 pv = p4[i];
            float4 lv = l4[i];
            int4   mv = m4[i];
            int o = i * 4;
            sKey[o + 0] = mv.x ? lv.x : kNEG;  sVal[o + 0] = mv.x ? pv.x : kNEG;
            sKey[o + 1] = mv.y ? lv.y : kNEG;  sVal[o + 1] = mv.y ? pv.y : kNEG;
            sKey[o + 2] = mv.z ? lv.z : kNEG;  sVal[o + 2] = mv.z ? pv.z : kNEG;
            sKey[o + 3] = mv.w ? lv.w : kNEG;  sVal[o + 3] = mv.w ? pv.w : kNEG;
        }
    }
    if (tid == 0) sK = 0;
    __syncthreads();

    // ---- bitonic sort, descending by key -----------------------------------
    // Thread t owns the contiguous block [8t, 8t+8). All stages with j<=4 stay
    // inside one block -> run them in registers (12 CEs), one LDS round-trip.
    const int b8 = tid * 8;
    float kr[8], vr[8];

    // Phase A: k = 2,4,8 are entirely register-local.
    #pragma unroll
    for (int i = 0; i < 8; ++i) { kr[i] = sKey[b8 + i]; vr[i] = sVal[b8 + i]; }
    #pragma unroll
    for (int k = 2; k <= 8; k <<= 1) {
        #pragma unroll
        for (int j = k >> 1; j >= 1; j >>= 1) {
            #pragma unroll
            for (int q = 0; q < 4; ++q) {
                int  li   = ((q & ~(j - 1)) << 1) | (q & (j - 1));
                bool desc = (((b8 + li) & k) == 0);
                float a = kr[li], b = kr[li + j];
                if ((a < b) == desc) {
                    kr[li] = b; kr[li + j] = a;
                    float t = vr[li]; vr[li] = vr[li + j]; vr[li + j] = t;
                }
            }
        }
    }
    #pragma unroll
    for (int i = 0; i < 8; ++i) { sKey[b8 + i] = kr[i]; sVal[b8 + i] = vr[i]; }
    __syncthreads();

    // Phase B: k = 16..2048; stages j>=8 in LDS, then j=4,2,1 in registers.
    for (int k = 16; k <= LROW; k <<= 1) {
        for (int j = k >> 1; j >= 8; j >>= 1) {
            for (int e = tid; e < LROW / 2; e += TPB) {
                int   i    = ((e & ~(j - 1)) << 1) | (e & (j - 1));
                int   ixj  = i | j;
                float a    = sKey[i], b = sKey[ixj];
                bool  desc = ((i & k) == 0);
                if ((a < b) == desc) {
                    sKey[i] = b; sKey[ixj] = a;
                    float va = sVal[i]; sVal[i] = sVal[ixj]; sVal[ixj] = va;
                }
            }
            __syncthreads();
        }
        #pragma unroll
        for (int i = 0; i < 8; ++i) { kr[i] = sKey[b8 + i]; vr[i] = sVal[b8 + i]; }
        #pragma unroll
        for (int j = 4; j >= 1; j >>= 1) {
            #pragma unroll
            for (int q = 0; q < 4; ++q) {
                int  li   = ((q & ~(j - 1)) << 1) | (q & (j - 1));
                bool desc = (((b8 + li) & k) == 0);
                float a = kr[li], b = kr[li + j];
                if ((a < b) == desc) {
                    kr[li] = b; kr[li + j] = a;
                    float t = vr[li]; vr[li] = vr[li + j]; vr[li + j] = t;
                }
            }
        }
        #pragma unroll
        for (int i = 0; i < 8; ++i) { sKey[b8 + i] = kr[i]; sVal[b8 + i] = vr[i]; }
        __syncthreads();
    }

    // ---- reverse (suffix) logcumsumexp -------------------------------------
    float locS[EPT];
    {
        float run = kNINF;
        #pragma unroll
        for (int i = EPT - 1; i >= 0; --i) {     // vr[] still holds sorted pm
            run     = lse2(run, vr[i]);
            locS[i] = run;
        }
        sScan[tid] = run;                        // chunk total
    }
    __syncthreads();
    // Hillis-Steele inclusive *suffix* scan of the 256 chunk totals
    for (int d = 1; d < TPB; d <<= 1) {
        float v = sScan[tid];
        float w = (tid + d < TPB) ? sScan[tid + d] : kNINF;
        __syncthreads();
        sScan[tid] = lse2(v, w);
        __syncthreads();
    }
    const float offs = (tid + 1 < TPB) ? sScan[tid + 1] : kNINF;

    // ---- terms + valid count (terms overwrite sKey; chunk-private) ---------
    {
        int kcount = 0;
        #pragma unroll
        for (int i = 0; i < EPT; ++i) {
            bool  valid = kr[i] > -5.0e8f;             // key != kNEG
            float lcs   = lse2(locS[i], offs);
            sKey[b8 + i] = valid ? (lcs - vr[i]) : 0.0f;
            kcount += valid ? 1 : 0;
        }
        atomicAdd(&sK, kcount);                        // ds_add_u32
    }
    __syncthreads();

    // ---- row sum of 2048 terms via WMMA against an all-ones B --------------
    // D[m][n] = sum_k A[m][k]; loss = sum over M of column 0, which lives in
    // lanes 0 (M=0..7) and 16 (M=8..15) of the 8-VGPR f32 accumulator.
    if (tid < 32) {                                    // wave 0, EXEC all-1s
#if __has_builtin(__builtin_amdgcn_wmma_f32_16x16x4_f32)
        v8f acc = {};
        v2f bones; bones.x = 1.0f; bones.y = 1.0f;
        for (int it = 0; it < LROW / 64; ++it) {       // 32 x (16x4 f32 tile)
            int p = it * 64 + tid * 2;
            v2f a; a.x = sKey[p]; a.y = sKey[p + 1];
            acc = __builtin_amdgcn_wmma_f32_16x16x4_f32(
                false, a, false, bones, (short)0, acc, false, false);
        }
#else
        v8f  acc = {};
        v16h bones;
        for (int i = 0; i < 16; ++i) bones[i] = (_Float16)1.0f;
        for (int it = 0; it < LROW / 512; ++it) {
            int  p = it * 512 + tid * 16;
            v16h a;
            for (int i = 0; i < 16; ++i) a[i] = (_Float16)sKey[p + i];
            acc = __builtin_amdgcn_wmma_f32_16x16x32_f16(
                false, a, false, bones, (short)0, acc, false, false);
        }
#endif
        if (tid == 0 || tid == 16) {
            float s = acc[0] + acc[1] + acc[2] + acc[3] +
                      acc[4] + acc[5] + acc[6] + acc[7];
            sPart[tid >> 4] = s;
        }
    }
    __syncthreads();

    if (tid == 0) {
        float loss = sPart[0] + sPart[1];
        int   kk   = sK;
        float lval = (kk > 1) ? loss : 0.0f;
        float nval = (kk > 1) ? 1.0f : 0.0f;
        if (atomicMode) {
            atomicAdd(&rowOut[0], lval);
            atomicAdd(&rowOut[1], nval);
        } else {
            rowOut[2 * row]     = lval;
            rowOut[2 * row + 1] = nval;
        }
    }
}

// Deterministic final reduction over per-row (loss, valid) pairs.
__global__ __launch_bounds__(1024) void listmle_reduce_kernel(
    const float* __restrict__ rowOut, int nRows, float* __restrict__ out)
{
    __shared__ float sT[1024];
    __shared__ float sN[1024];
    int   tid = threadIdx.x;
    float t = 0.0f, n = 0.0f;
    const float2* ro2 = (const float2*)rowOut;
    for (int r = tid; r < nRows; r += 1024) {
        float2 p = ro2[r];
        t += p.x;
        n += p.y;
    }
    sT[tid] = t; sN[tid] = n;
    __syncthreads();
    for (int s = 512; s > 0; s >>= 1) {
        if (tid < s) { sT[tid] += sT[tid + s]; sN[tid] += sN[tid + s]; }
        __syncthreads();
    }
    if (tid == 0) out[0] = (sN[0] > 0.0f) ? (sT[0] / sN[0]) : 0.0f;
}

// Fallback path helpers (only used if ws_size is tiny)
__global__ void listmle_zero2_kernel(float* acc) { acc[0] = 0.0f; acc[1] = 0.0f; }
__global__ void listmle_final_atomic_kernel(const float* acc, float* out) {
    out[0] = (acc[1] > 0.0f) ? (acc[0] / acc[1]) : 0.0f;
}

extern "C" void kernel_launch(void* const* d_in, const int* in_sizes, int n_in,
                              void* d_out, int out_size, void* d_ws, size_t ws_size,
                              hipStream_t stream) {
    const float* preds  = (const float*)d_in[0];
    const float* labels = (const float*)d_in[1];
    const int*   mask   = (const int*)d_in[2];
    float*       out    = (float*)d_out;
    float*       ws     = (float*)d_ws;
    const int    B      = in_sizes[0] / LROW;   // 8192

    const size_t need = (size_t)B * 2 * sizeof(float);
    if (ws_size >= need) {
        listmle_row_kernel<<<B, TPB, 0, stream>>>(preds, labels, mask, ws, 0);
        listmle_reduce_kernel<<<1, 1024, 0, stream>>>(ws, B, out);
    } else {
        listmle_zero2_kernel<<<1, 1, 0, stream>>>(ws);
        listmle_row_kernel<<<B, TPB, 0, stream>>>(preds, labels, mask, ws, 1);
        listmle_final_atomic_kernel<<<1, 1, 0, stream>>>(ws, out);
    }
}
